// Attention_63548336111918
// MI455X (gfx1250) — matrix operand order repaired and checked by reading, not compiled, not run on hardware
//
#include <hip/hip_runtime.h>
#include <hip/hip_bf16.h>

// ---------------------------------------------------------------------------
// softmax(q @ k^T * d^-0.5) for q[16,2048,512], k[16,2048,512] fp32.
// bf16x3 split-GEMM on v_wmma_f32_16x16x32_bf16, fp32 accumulate.
// Each wave: 32 rows x 128 cols of scores in registers (2 row-tiles x 8
// col-tiles); B operands shared across the two row-tiles to halve L2 traffic.
// Workgroup: 16 waves = 32 rows x 2048 cols; full-row softmax in-registers
// with wave shuffle + small LDS cross-wave reduction.
// ---------------------------------------------------------------------------

typedef __attribute__((ext_vector_type(16))) __bf16 v16bf;
typedef __attribute__((ext_vector_type(8)))  float  v8f;

#define BATCH   16
#define NQ      2048
#define MK      2048
#define DIM     512
#define ROWS    32            // q rows per workgroup (= rows per wave)
#define THREADS 512           // 16 wave32s
#define NCG     16            // column groups (one per wave), 128 cols each
#define SCALE   0.0441941738241592f   // 1/sqrt(512)

union BV16 { unsigned short u[16]; v16bf v; };

__device__ __forceinline__ unsigned short f32_to_bf16_rne(float x) {
  unsigned int u = __float_as_uint(x);
  u += 0x7FFFu + ((u >> 16) & 1u);
  return (unsigned short)(u >> 16);
}

__device__ __forceinline__ void bfsplit(float v, unsigned short& h, unsigned short& l) {
  h = f32_to_bf16_rne(v);
  float hf = __uint_as_float((unsigned int)h << 16);
  l = f32_to_bf16_rne(v - hf);
}

// ---------------------------------------------------------------------------
// Prep: split k fp32 -> (hi, lo) bf16 arrays in workspace (elementwise).
// ---------------------------------------------------------------------------
__global__ __launch_bounds__(256)
void ksplit_kernel(const float* __restrict__ src,
                   unsigned short* __restrict__ hi,
                   unsigned short* __restrict__ lo) {
  size_t i = ((size_t)blockIdx.x * blockDim.x + threadIdx.x) * 4;
  float4 v = *(const float4*)(src + i);
  ushort4 h, l;
  bfsplit(v.x, h.x, l.x);
  bfsplit(v.y, h.y, l.y);
  bfsplit(v.z, h.z, l.z);
  bfsplit(v.w, h.w, l.w);
  *(ushort4*)(hi + i) = h;
  *(ushort4*)(lo + i) = l;
}

// ---------------------------------------------------------------------------
// Main kernel. grid = (NQ/ROWS, BATCH), block = 512.
// wave w = column group cg (128 cols); every wave covers all 32 rows.
// ---------------------------------------------------------------------------
template <bool SPLITK>
__global__ __launch_bounds__(THREADS)
void attn_softmax_kernel(const float* __restrict__ q,
                         const float* __restrict__ kf,
                         const unsigned short* __restrict__ khi,
                         const unsigned short* __restrict__ klo,
                         float* __restrict__ out) {
  extern __shared__ char smem[];
  unsigned short* qhi   = (unsigned short*)smem;            // ROWS*DIM bf16 (32 KB)
  unsigned short* qlo   = qhi + ROWS * DIM;                 // ROWS*DIM bf16 (32 KB)
  float* redmx   = (float*)(qlo + ROWS * DIM);              // [16 cg][32 rows]
  float* redsm   = redmx + NCG * ROWS;                      // [16 cg][32 rows]
  float* redfinm = redsm + NCG * ROWS;                      // [32 rows]
  float* redfins = redfinm + ROWS;                          // [32 rows]

  const int tid = threadIdx.x;
  const int bt  = blockIdx.y;
  const int nb  = blockIdx.x * ROWS;

  // ---- stage q tile into LDS as bf16 hi/lo (coalesced: tid == d index) ----
  for (int row = 0; row < ROWS; ++row) {
    float v = q[((size_t)bt * NQ + nb + row) * DIM + tid];
    unsigned short h, l;
    bfsplit(v, h, l);
    qhi[row * DIM + tid] = h;
    qlo[row * DIM + tid] = l;
  }
  __syncthreads();

  const int lane  = tid & 31;
  const int cg    = tid >> 5;        // column group = wave id
  const int lhalf = lane >> 4;       // 0: lanes 0-15, 1: lanes 16-31
  const int l16   = lane & 15;

  v8f acc[2][8];
#pragma unroll
  for (int ra = 0; ra < 2; ++ra)
#pragma unroll
    for (int t = 0; t < 8; ++t) { v8f z = {}; acc[ra][t] = z; }

  for (int kc = 0; kc < DIM / 32; ++kc) {
    // A operands for both row-tiles from LDS.
    // A k order used here (contiguous, B matches): lanes 0-15 -> M=lane, K in {0..15};
    //                  lanes 16-31 -> M=lane-16, K in {16..31}.
    const int ka = kc * 32 + lhalf * 16;
    BV16 ah[2], al[2];
#pragma unroll
    for (int ra = 0; ra < 2; ++ra) {
      const unsigned short* qhr = qhi + (ra * 16 + l16) * DIM;
      const unsigned short* qlr = qlo + (ra * 16 + l16) * DIM;
      *(uint4*)&ah[ra].u[0] = *(const uint4*)&qhr[ka];
      *(uint4*)&ah[ra].u[8] = *(const uint4*)&qhr[ka + 8];
      *(uint4*)&al[ra].u[0] = *(const uint4*)&qlr[ka];
      *(uint4*)&al[ra].u[8] = *(const uint4*)&qlr[ka + 8];
    }

    // B operand: lane -> k row (N), contiguous 16 K elems per lane.
    const int kb = kc * 32 + lhalf * 16;

#pragma unroll
    for (int t = 0; t < 8; ++t) {
      const int ncol = cg * 128 + t * 16 + l16;
      const size_t kofs = ((size_t)bt * MK + ncol) * DIM + kb;
      BV16 bh, bl;
      if constexpr (SPLITK) {
        *(uint4*)&bh.u[0] = *(const uint4*)(khi + kofs);
        *(uint4*)&bh.u[8] = *(const uint4*)(khi + kofs + 8);
        *(uint4*)&bl.u[0] = *(const uint4*)(klo + kofs);
        *(uint4*)&bl.u[8] = *(const uint4*)(klo + kofs + 8);
      } else {
        const float4* kp = (const float4*)(kf + kofs);
        float4 c0 = kp[0], c1 = kp[1], c2 = kp[2], c3 = kp[3];
        const float vv[16] = {c0.x,c0.y,c0.z,c0.w, c1.x,c1.y,c1.z,c1.w,
                              c2.x,c2.y,c2.z,c2.w, c3.x,c3.y,c3.z,c3.w};
#pragma unroll
        for (int j = 0; j < 16; ++j) bfsplit(vv[j], bh.u[j], bl.u[j]);
      }
      // bf16x3 for both row-tiles; B reused across row-tiles.
#pragma unroll
      for (int ra = 0; ra < 2; ++ra) {
        acc[ra][t] = __builtin_amdgcn_wmma_f32_16x16x32_bf16(false, ah[ra].v, false, bh.v,
                                                             (short)0, acc[ra][t], false, false);
        acc[ra][t] = __builtin_amdgcn_wmma_f32_16x16x32_bf16(false, al[ra].v, false, bh.v,
                                                             (short)0, acc[ra][t], false, false);
        acc[ra][t] = __builtin_amdgcn_wmma_f32_16x16x32_bf16(false, ah[ra].v, false, bl.v,
                                                             (short)0, acc[ra][t], false, false);
      }
    }
  }

  // ---- softmax over the full row (2048 cols) ----
  // C/D layout: lanes 0-15 hold rows i (VGPR i), lanes 16-31 rows i+8, col = l16.
  float rmax[2][8];
#pragma unroll
  for (int ra = 0; ra < 2; ++ra)
#pragma unroll
    for (int i = 0; i < 8; ++i) {
      float m = acc[ra][0][i];
#pragma unroll
      for (int t = 1; t < 8; ++t) m = fmaxf(m, acc[ra][t][i]);
      rmax[ra][i] = m;
    }
#pragma unroll
  for (int off = 1; off < 16; off <<= 1)
#pragma unroll
    for (int ra = 0; ra < 2; ++ra)
#pragma unroll
      for (int i = 0; i < 8; ++i)
        rmax[ra][i] = fmaxf(rmax[ra][i], __shfl_xor(rmax[ra][i], off, 32));
  if (l16 < 8) {
#pragma unroll
    for (int ra = 0; ra < 2; ++ra)
      redmx[cg * ROWS + ra * 16 + lhalf * 8 + l16] = rmax[ra][l16];
  }
  __syncthreads();

  // lane-parallel cross-wave max: wave 0, one lane per row
  if (tid < ROWS) {
    float m = redmx[tid];
#pragma unroll
    for (int c = 1; c < NCG; ++c) m = fmaxf(m, redmx[c * ROWS + tid]);
    redfinm[tid] = m;
  }
  __syncthreads();

  float rmaxf[2][8];
#pragma unroll
  for (int ra = 0; ra < 2; ++ra)
#pragma unroll
    for (int i = 0; i < 8; ++i)
      rmaxf[ra][i] = redfinm[ra * 16 + lhalf * 8 + i];

  float rsum[2][8];
#pragma unroll
  for (int ra = 0; ra < 2; ++ra)
#pragma unroll
    for (int i = 0; i < 8; ++i) rsum[ra][i] = 0.0f;
#pragma unroll
  for (int ra = 0; ra < 2; ++ra)
#pragma unroll
    for (int t = 0; t < 8; ++t)
#pragma unroll
      for (int i = 0; i < 8; ++i) {
        float e = __expf((acc[ra][t][i] - rmaxf[ra][i]) * SCALE);
        acc[ra][t][i] = e;
        rsum[ra][i] += e;
      }
#pragma unroll
  for (int off = 1; off < 16; off <<= 1)
#pragma unroll
    for (int ra = 0; ra < 2; ++ra)
#pragma unroll
      for (int i = 0; i < 8; ++i)
        rsum[ra][i] += __shfl_xor(rsum[ra][i], off, 32);
  if (l16 < 8) {
#pragma unroll
    for (int ra = 0; ra < 2; ++ra)
      redsm[cg * ROWS + ra * 16 + lhalf * 8 + l16] = rsum[ra][l16];
  }
  __syncthreads();

  if (tid < ROWS) {
    float s = 0.0f;
#pragma unroll
    for (int c = 0; c < NCG; ++c) s += redsm[c * ROWS + tid];
    redfins[tid] = s;
  }
  __syncthreads();

  float rinv[2][8];
#pragma unroll
  for (int ra = 0; ra < 2; ++ra)
#pragma unroll
    for (int i = 0; i < 8; ++i)
      rinv[ra][i] = 1.0f / redfins[ra * 16 + lhalf * 8 + i];

  // ---- write normalized probabilities ----
#pragma unroll
  for (int ra = 0; ra < 2; ++ra)
#pragma unroll
    for (int t = 0; t < 8; ++t) {
      const int col = cg * 128 + t * 16 + l16;
#pragma unroll
      for (int i = 0; i < 8; ++i) {
        const int row = nb + ra * 16 + lhalf * 8 + i;
        out[((size_t)bt * NQ + row) * MK + col] = acc[ra][t][i] * rinv[ra][i];
      }
    }
}

// ---------------------------------------------------------------------------
extern "C" void kernel_launch(void* const* d_in, const int* in_sizes, int n_in,
                              void* d_out, int out_size, void* d_ws, size_t ws_size,
                              hipStream_t stream) {
  const float* q = (const float*)d_in[0];
  const float* k = (const float*)d_in[1];
  float* out = (float*)d_out;

  const size_t kelems = (size_t)BATCH * MK * DIM;               // 16M
  const size_t wsneed = kelems * 2 * sizeof(unsigned short);    // hi+lo = 64 MB
  const size_t smem   = (size_t)ROWS * DIM * 2 * 2              // q hi/lo
                      + 2 * (NCG * ROWS) * 4                    // redmx/redsm
                      + 2 * ROWS * 4;                           // redfinm/redfins

  dim3 grid(NQ / ROWS, BATCH);
  dim3 block(THREADS);

  if (ws_size >= wsneed) {
    unsigned short* khi = (unsigned short*)d_ws;
    unsigned short* klo = khi + kelems;
    const int sthreads = 256;
    const int sblocks  = (int)(kelems / 4 / sthreads);          // 16384
    ksplit_kernel<<<sblocks, sthreads, 0, stream>>>(k, khi, klo);
    attn_softmax_kernel<true><<<grid, block, smem, stream>>>(q, k, khi, klo, out);
  } else {
    attn_softmax_kernel<false><<<grid, block, smem, stream>>>(q, k, nullptr, nullptr, out);
  }
}
